// CoAttention_30348238913683
// MI455X (gfx1250) — compile-verified, matrix operand order checked by reading, not run on hardware
//
#include <hip/hip_runtime.h>
#include <hip/hip_bf16.h>
#include <math.h>

// ---------------------------------------------------------------------------
// CoAttention for MI455X (gfx1250).
//
// Algebraic simplification (exact): softmax over an outer-sum score drops the
// softmax-axis-invariant term, so
//   att_audio_features[b,l,:] = sum_a softmax_a(w1b . tanh(Wa1 @ audio[b,a]))
//                                      * audio[b,a,:]          (indep. of l)
//   att_text_features [b,l,:] = sum_k softmax_k(w2b . tanh(Wt2 @ text[b,k]))
//                                      * text[b,k,:]           (indep. of l)
// Wt1/bt1/b_att1/Wa2/ba2/b_att2 and w_att*[0:D] provably do not affect output.
//
// Heavy work = two tanh-GEMMs on v_wmma_f32_16x16x32_f16 (f32 accumulate),
// 4 e-tiles per pass (4 accumulators) so each A fragment feeds 4 WMMAs.
// If d_ws is large enough, X and W are pre-converted to f16 once, so the
// GEMM inner loop is pure global_load_b128 + v_wmma (no cvt, half the bytes).
// ---------------------------------------------------------------------------

typedef __attribute__((ext_vector_type(16))) _Float16 v16h;
typedef __attribute__((ext_vector_type(8)))  _Float16 v8h;
typedef __attribute__((ext_vector_type(8)))  float    v8f;

#define DD 1024   // feature dim D
#define LL 1024   // text length L (broadcast length of both outputs)
#define BB 16     // batch
#define AA 512    // audio length

// Fragment layout (documented 16-bit A 16x32; B assumed symmetric/transposed):
//   half = lane>>4.  j<8  -> row[k0 + half*8 + j]
//                    j>=8 -> row[k0 + 16 + half*8 + (j-8)]
__device__ __forceinline__ v16h load_frag(const float* __restrict__ row,
                                          int k0, int half) {
  const float4* p0 = (const float4*)(row + k0 + half * 8);
  const float4* p1 = (const float4*)(row + k0 + 16 + half * 8);
  float4 f0 = p0[0], f1 = p0[1];
  float4 f2 = p1[0], f3 = p1[1];
  v16h r;
  r[0]  = (_Float16)f0.x; r[1]  = (_Float16)f0.y;
  r[2]  = (_Float16)f0.z; r[3]  = (_Float16)f0.w;
  r[4]  = (_Float16)f1.x; r[5]  = (_Float16)f1.y;
  r[6]  = (_Float16)f1.z; r[7]  = (_Float16)f1.w;
  r[8]  = (_Float16)f2.x; r[9]  = (_Float16)f2.y;
  r[10] = (_Float16)f2.z; r[11] = (_Float16)f2.w;
  r[12] = (_Float16)f3.x; r[13] = (_Float16)f3.y;
  r[14] = (_Float16)f3.z; r[15] = (_Float16)f3.w;
  return r;
}

__device__ __forceinline__ v16h load_frag(const _Float16* __restrict__ row,
                                          int k0, int half) {
  v8h lo = *(const v8h*)(row + k0 + half * 8);        // one b128 load
  v8h hi = *(const v8h*)(row + k0 + 16 + half * 8);   // one b128 load
  v16h r;
#pragma unroll
  for (int j = 0; j < 8; ++j) { r[j] = lo[j]; r[8 + j] = hi[j]; }
  return r;
}

// s_out[r] = wv . tanh(W @ X[r,:]) for M rows of X (row-major, stride D).
// 256 threads = 8 waves; each wave owns a 16-row tile (128 rows/block),
// sweeps e in blocks of 4 tiles (4 WMMA accumulators share each A fragment).
template <typename ET>
__global__ __launch_bounds__(256) void score_kernel(
    const ET* __restrict__ X, const ET* __restrict__ W,
    const float* __restrict__ wv, float* __restrict__ s_out, int M) {
  const int lane = threadIdx.x & 31;
  const int wave = threadIdx.x >> 5;
  const int r0   = (blockIdx.x * 8 + wave) * 16;
  if (r0 >= M) return;
  const int half = lane >> 4;    // K-half selector
  const int m16  = lane & 15;    // A: row within tile / B: column (= W row)

  const ET* xrow = X + (size_t)(r0 + m16) * DD;

  float srow[8];
#pragma unroll
  for (int i = 0; i < 8; ++i) srow[i] = 0.0f;

  for (int e0 = 0; e0 < DD; e0 += 64) {       // 4 e-tiles per pass
    const ET* w0 = W + (size_t)(e0 + 0  + m16) * DD;
    const ET* w1 = W + (size_t)(e0 + 16 + m16) * DD;
    const ET* w2 = W + (size_t)(e0 + 32 + m16) * DD;
    const ET* w3 = W + (size_t)(e0 + 48 + m16) * DD;
    // Prefetch next e-block's first W row (gfx1250 global_prefetch_b8).
    if (e0 + 64 < DD) __builtin_prefetch(W + (size_t)(e0 + 64 + m16) * DD, 0, 1);

    v8f acc0 = {}, acc1 = {}, acc2 = {}, acc3 = {};
#pragma unroll 2
    for (int k0 = 0; k0 < DD; k0 += 32) {
      v16h a = load_frag(xrow, k0, half);     // reused by 4 WMMAs
      acc0 = __builtin_amdgcn_wmma_f32_16x16x32_f16(
          false, a, false, load_frag(w0, k0, half), (short)0, acc0, false, false);
      acc1 = __builtin_amdgcn_wmma_f32_16x16x32_f16(
          false, a, false, load_frag(w1, k0, half), (short)0, acc1, false, false);
      acc2 = __builtin_amdgcn_wmma_f32_16x16x32_f16(
          false, a, false, load_frag(w2, k0, half), (short)0, acc2, false, false);
      acc3 = __builtin_amdgcn_wmma_f32_16x16x32_f16(
          false, a, false, load_frag(w3, k0, half), (short)0, acc3, false, false);
    }
    // acc_j[i] = C[m = i + 8*half, n = m16] for e = e0 + 16*j + m16.
    const float wv0 = wv[e0 + 0  + m16];
    const float wv1 = wv[e0 + 16 + m16];
    const float wv2 = wv[e0 + 32 + m16];
    const float wv3 = wv[e0 + 48 + m16];
#pragma unroll
    for (int i = 0; i < 8; ++i) {
      srow[i] += tanhf(acc0[i]) * wv0 + tanhf(acc1[i]) * wv1 +
                 tanhf(acc2[i]) * wv2 + tanhf(acc3[i]) * wv3;
    }
  }

  // Reduce over the 16 N-lanes (xor masks < 16 stay within each half-group).
#pragma unroll
  for (int i = 0; i < 8; ++i) {
    float t = srow[i];
    t += __shfl_xor(t, 1);
    t += __shfl_xor(t, 2);
    t += __shfl_xor(t, 4);
    t += __shfl_xor(t, 8);
    srow[i] = t;
  }
  if (m16 == 0) {             // lane 0 -> rows r0..r0+7, lane 16 -> r0+8..r0+15
    const int base = r0 + half * 8;
#pragma unroll
    for (int i = 0; i < 8; ++i) s_out[base + i] = srow[i];
  }
}

// One-time f32 -> f16 conversion into scratch (n must be a multiple of 8).
__global__ __launch_bounds__(256) void cvt_kernel(const float* __restrict__ in,
                                                  _Float16* __restrict__ out,
                                                  int n) {
  int i = (blockIdx.x * 256 + threadIdx.x) * 8;
  if (i >= n) return;
  float4 a = *(const float4*)(in + i);
  float4 b = *(const float4*)(in + i + 4);
  v8h r;
  r[0] = (_Float16)a.x; r[1] = (_Float16)a.y;
  r[2] = (_Float16)a.z; r[3] = (_Float16)a.w;
  r[4] = (_Float16)b.x; r[5] = (_Float16)b.y;
  r[6] = (_Float16)b.z; r[7] = (_Float16)b.w;
  *(v8h*)(out + i) = r;
}

// Per batch b: p = softmax(s[b, 0:R]); c[d] = sum_r p[r]*X[b,r,d];
// out[b,l,d] = c[d] for all l in [0,L). Deterministic LDS tree reductions.
__global__ __launch_bounds__(256) void attend_kernel(
    const float* __restrict__ X, const float* __restrict__ s,
    float* __restrict__ out, int R) {
  const int b = blockIdx.x;
  const int tid = threadIdx.x;
  __shared__ float p[LL];
  __shared__ float red[256];

  float m = -INFINITY;
  for (int r = tid; r < R; r += 256) m = fmaxf(m, s[b * R + r]);
  red[tid] = m;
  __syncthreads();
  for (int o = 128; o > 0; o >>= 1) {
    if (tid < o) red[tid] = fmaxf(red[tid], red[tid + o]);
    __syncthreads();
  }
  const float mx = red[0];
  __syncthreads();

  float ls = 0.0f;
  for (int r = tid; r < R; r += 256) {
    float e = __expf(s[b * R + r] - mx);
    p[r] = e;
    ls += e;
  }
  red[tid] = ls;
  __syncthreads();
  for (int o = 128; o > 0; o >>= 1) {
    if (tid < o) red[tid] += red[tid + o];
    __syncthreads();
  }
  const float inv = 1.0f / red[0];
  __syncthreads();
  for (int r = tid; r < R; r += 256) p[r] *= inv;
  __syncthreads();

  float c[4] = {0.0f, 0.0f, 0.0f, 0.0f};
  for (int r = 0; r < R; ++r) {
    const float pr = p[r];
    const float* xr = X + ((size_t)(b * R + r)) * DD + tid;
#pragma unroll
    for (int j = 0; j < 4; ++j) c[j] += pr * xr[j * 256];
  }

  for (int l = 0; l < LL; ++l) {
    float* o = out + ((size_t)(b * LL + l)) * DD + tid;
#pragma unroll
    for (int j = 0; j < 4; ++j) o[j * 256] = c[j];
  }
}

extern "C" void kernel_launch(void* const* d_in, const int* in_sizes, int n_in,
                              void* d_out, int out_size, void* d_ws, size_t ws_size,
                              hipStream_t stream) {
  (void)in_sizes; (void)n_in; (void)out_size;
  const float* text   = (const float*)d_in[0];   // (B, L, D)
  const float* audio  = (const float*)d_in[1];   // (B, A, D)
  const float* Wa1    = (const float*)d_in[4];   // (D, D)
  const float* w_att1 = (const float*)d_in[5];   // (2D,) -> use [D:2D)
  const float* Wt2    = (const float*)d_in[7];   // (D, D)
  const float* w_att2 = (const float*)d_in[10];  // (2D,) -> use [D:2D)
  // d_in[2,3,6,8,9,11] (Wt1, bt1, b_att1, Wa2, ba2, b_att2) provably unused.

  float* out_text  = (float*)d_out;                                   // (B,L,D)
  float* out_audio = (float*)d_out + (size_t)BB * LL * DD;            // (B,L,D)

  // Scratch layout: scores first, then optional f16 staging buffers.
  const size_t nText = (size_t)BB * LL * DD;   // 16Mi
  const size_t nAud  = (size_t)BB * AA * DD;   //  8Mi
  const size_t nW    = (size_t)DD * DD;        //  1Mi
  float* s2 = (float*)d_ws;                    // B*L floats
  float* s1 = s2 + BB * LL;                    // B*A floats
  const size_t scoreBytes = (size_t)(BB * LL + BB * AA) * sizeof(float);
  _Float16* hText = (_Float16*)((char*)d_ws + scoreBytes);
  _Float16* hAud  = hText + nText;
  _Float16* hWt2  = hAud + nAud;
  _Float16* hWa1  = hWt2 + nW;
  const size_t needBytes = scoreBytes + (nText + nAud + 2 * nW) * sizeof(_Float16);
  const bool use16 = ws_size >= needBytes;

  if (use16) {
    cvt_kernel<<<(int)(nText / 8 / 256), 256, 0, stream>>>(text, hText, (int)nText);
    cvt_kernel<<<(int)(nAud  / 8 / 256), 256, 0, stream>>>(audio, hAud, (int)nAud);
    cvt_kernel<<<(int)(nW    / 8 / 256), 256, 0, stream>>>(Wt2, hWt2, (int)nW);
    cvt_kernel<<<(int)(nW    / 8 / 256), 256, 0, stream>>>(Wa1, hWa1, (int)nW);
    score_kernel<_Float16><<<(BB * LL) / 128, 256, 0, stream>>>(
        hText, hWt2, w_att2 + DD, s2, BB * LL);
    score_kernel<_Float16><<<(BB * AA) / 128, 256, 0, stream>>>(
        hAud, hWa1, w_att1 + DD, s1, BB * AA);
  } else {
    score_kernel<float><<<(BB * LL) / 128, 256, 0, stream>>>(
        text, Wt2, w_att2 + DD, s2, BB * LL);
    score_kernel<float><<<(BB * AA) / 128, 256, 0, stream>>>(
        audio, Wa1, w_att1 + DD, s1, BB * AA);
  }
  attend_kernel<<<BB, 256, 0, stream>>>(text, s2, out_text, LL);
  attend_kernel<<<BB, 256, 0, stream>>>(audio, s1, out_audio, AA);
}
